// SelfAttention_74380243632309
// MI455X (gfx1250) — matrix operand order repaired and checked by reading, not compiled, not run on hardware
//
#include <hip/hip_runtime.h>
#include <hip/hip_bf16.h>

#define BATCH  4
#define CDIM   256
#define NTOK   4096
#define NGRP   32
#define CPG    (CDIM / NGRP)
#define GN_EPS 1e-5f
#define SCALE  0.0625f   // CDIM^-0.5

#define LDSP        40   // padded LDS row stride in halfs (80 B -> bank spread)
#define PANEL_BYTES 64   // 32 halfs of K per panel row

typedef __attribute__((ext_vector_type(16))) _Float16 v16h;
typedef __attribute__((ext_vector_type(8)))  _Float16 v8h;
typedef __attribute__((ext_vector_type(8)))  float    v8f;

union AFrag { v16h v; v8h h[2]; };

// ----------------------------------------------------------- async LDS copy
// Per-lane 16B async copy: LDS[lds_off] = MEM[gaddr].  Tracked by ASYNCcnt.
__device__ __forceinline__ void async_cp16(uint32_t lds_off, uint64_t gaddr) {
  asm volatile("global_load_async_to_lds_b128 %0, %1, off"
               :: "v"(lds_off), "v"(gaddr)
               : "memory");
}
__device__ __forceinline__ void wait_async0() {
  asm volatile("s_wait_asynccnt 0x0" ::: "memory");
}

// A (16x32 f16): lane l holds row l&15; e=0..7 -> k=(l>>4)*8+e,
// e=8..15 -> k=16+(l>>4)*8+(e-8).  Two 16B reads.
__device__ __forceinline__ v16h lds_a_frag(const _Float16* p) {
  AFrag f;
  f.h[0] = *(const v8h*)(p);
  f.h[1] = *(const v8h*)(p + 16);
  return f.v;
}

__device__ __forceinline__ v8f wmma16(v16h a, v16h b, v8f c) {
  return __builtin_amdgcn_wmma_f32_16x16x32_f16(false, a, false, b, (short)0, c,
                                                false, false);
}

// ------------------------------------------------------- staged NT-GEMM core
// Block tile 128(M) x 128(N), K panels of 32, double-buffered async staging.
// 8 waves arranged 4(M) x 2(N); each wave computes a 32x64 slab (8 accums).
// D[m][n] += sum_k A[m*lda+k] * Bt[n*ldb+k]
__device__ __forceinline__ void gemm_core_128(
    const _Float16* __restrict__ A, int lda,
    const _Float16* __restrict__ Bt, int ldb,
    int K, int m0, int n0, v8f acc[2][4]) {
  __shared__ __align__(16) _Float16 sA[2][128 * LDSP];
  __shared__ __align__(16) _Float16 sB[2][128 * LDSP];

  const int t    = threadIdx.x;
  const int lane = t & 31, w = t >> 5;
  const int wm   = w >> 1, wn = w & 1;
  const int hlf  = lane >> 4, l15 = lane & 15;

  // Each thread stages 2x16B of A and 2x16B of B per panel.
  uint64_t gA[2], gB[2];
  uint32_t oA[2][2], oB[2][2];
  for (int j = 0; j < 2; ++j) {
    const int chunk = t + 256 * j;           // 512 chunks of 16B per panel
    const int row = chunk >> 2;              // 0..127
    const int q   = (chunk & 3) * 8;         // half offset within 32-wide row
    gA[j] = (uint64_t)(A  + (size_t)(m0 + row) * lda + q);
    gB[j] = (uint64_t)(Bt + (size_t)(n0 + row) * ldb + q);
    for (int u = 0; u < 2; ++u) {
      oA[j][u] = (uint32_t)(size_t)&sA[u][row * LDSP + q];
      oB[j][u] = (uint32_t)(size_t)&sB[u][row * LDSP + q];
    }
  }

  const int P = K >> 5;
  // prologue: panel 0 -> buffer 0
  for (int j = 0; j < 2; ++j) {
    async_cp16(oA[j][0], gA[j]);
    async_cp16(oB[j][0], gB[j]);
    gA[j] += PANEL_BYTES; gB[j] += PANEL_BYTES;
  }

  for (int i = 0; i < P; ++i) {
    wait_async0();        // my async writes into buf (i&1) complete
    __syncthreads();      // everyone's complete; prev-panel readers done
    const int cur = i & 1;
    if (i + 1 < P) {
      const int nxt = cur ^ 1;
      for (int j = 0; j < 2; ++j) {
        async_cp16(oA[j][nxt], gA[j]);
        async_cp16(oB[j][nxt], gB[j]);
        gA[j] += PANEL_BYTES; gB[j] += PANEL_BYTES;
      }
    }
    const _Float16* pa = &sA[cur][(wm * 32 + l15) * LDSP + hlf * 8];
    const _Float16* pb = &sB[cur][(wn * 64 + l15) * LDSP + hlf * 8];
    v16h af0 = lds_a_frag(pa);
    v16h af1 = lds_a_frag(pa + 16 * LDSP);
    v16h bf[4];
    for (int ni = 0; ni < 4; ++ni)
      bf[ni] = lds_a_frag(pb + ni * 16 * LDSP);
    for (int ni = 0; ni < 4; ++ni) {
      acc[0][ni] = wmma16(af0, bf[ni], acc[0][ni]);
      acc[1][ni] = wmma16(af1, bf[ni], acc[1][ni]);
    }
  }
}

// ---------------------------------------------------------------- GroupNorm
__global__ void groupnorm_kernel(const float* __restrict__ x,
                                 const float* __restrict__ gamma,
                                 const float* __restrict__ beta,
                                 _Float16* __restrict__ hT) {
  const int b = blockIdx.x / NGRP;
  const int g = blockIdx.x % NGRP;
  const int t = threadIdx.x;
  const int M = CPG * NTOK;
  const float* base = x + ((size_t)b * CDIM + g * CPG) * NTOK;

  float ls = 0.f, lq = 0.f;
  for (int i = t; i < M; i += 256) {
    float v = base[i];
    ls += v; lq += v * v;
  }
  __shared__ float s0[256], s1[256];
  s0[t] = ls; s1[t] = lq;
  __syncthreads();
  for (int o = 128; o > 0; o >>= 1) {
    if (t < o) { s0[t] += s0[t + o]; s1[t] += s1[t + o]; }
    __syncthreads();
  }
  const float mean = s0[0] / (float)M;
  const float var  = s1[0] / (float)M - mean * mean;
  const float inv  = rsqrtf(var + GN_EPS);

  for (int i = t; i < M; i += 256) {
    const int cl = i / NTOK, n = i % NTOK;
    const int c  = g * CPG + cl;
    const float h = (base[i] - mean) * inv * gamma[c] + beta[c];
    hT[((size_t)b * NTOK + n) * CDIM + c] = (_Float16)h;
  }
}

__global__ void f32_to_f16(const float* __restrict__ src,
                           _Float16* __restrict__ dst, int n) {
  int i = blockIdx.x * 256 + threadIdx.x;
  if (i < n) dst[i] = (_Float16)src[i];
}

// ------------------------------------------------------------------ QKV GEMM
__global__ __launch_bounds__(256) void qkv_gemm(
    const _Float16* __restrict__ Wq, const _Float16* __restrict__ hT,
    const float* __restrict__ qkvb, _Float16* __restrict__ qT,
    _Float16* __restrict__ kT, _Float16* __restrict__ vB) {
  const int b  = blockIdx.z;
  const int m0 = blockIdx.y * 128, n0 = blockIdx.x * 128;
  v8f z = {};
  v8f acc[2][4];
  for (int mi = 0; mi < 2; ++mi) for (int ni = 0; ni < 4; ++ni) acc[mi][ni] = z;

  gemm_core_128(Wq, CDIM, hT + (size_t)b * NTOK * CDIM, CDIM, CDIM, m0, n0, acc);

  const int lane = threadIdx.x & 31, w = threadIdx.x >> 5;
  const int hlf = lane >> 4, l15 = lane & 15;
  const int gm = m0 + (w >> 1) * 32, gn = n0 + (w & 1) * 64;
  for (int mi = 0; mi < 2; ++mi)
    for (int r = 0; r < 8; ++r) {
      const int d = gm + mi * 16 + r + hlf * 8;
      const int sec = d >> 8, c = d & 255;
      const float bias = qkvb[d];
      for (int ni = 0; ni < 4; ++ni) {
        const int n = gn + ni * 16 + l15;
        const float val = acc[mi][ni][r] + bias;
        if (sec == 0)      qT[((size_t)b * NTOK + n) * CDIM + c] = (_Float16)val;
        else if (sec == 1) kT[((size_t)b * NTOK + n) * CDIM + c] = (_Float16)val;
        else               vB[((size_t)b * CDIM + c) * NTOK + n] = (_Float16)val;
      }
    }
}

// --------------------------------------------------------------- Scores GEMM
__global__ __launch_bounds__(256) void scores_gemm(
    const _Float16* __restrict__ qT, const _Float16* __restrict__ kT,
    _Float16* __restrict__ attn) {
  const int b  = blockIdx.z;
  const int m0 = blockIdx.y * 128, n0 = blockIdx.x * 128;
  v8f z = {};
  v8f acc[2][4];
  for (int mi = 0; mi < 2; ++mi) for (int ni = 0; ni < 4; ++ni) acc[mi][ni] = z;

  gemm_core_128(qT + (size_t)b * NTOK * CDIM, CDIM,
                kT + (size_t)b * NTOK * CDIM, CDIM, CDIM, m0, n0, acc);

  const int lane = threadIdx.x & 31, w = threadIdx.x >> 5;
  const int hlf = lane >> 4, l15 = lane & 15;
  const int gm = m0 + (w >> 1) * 32, gn = n0 + (w & 1) * 64;
  _Float16* base = attn + (size_t)b * NTOK * NTOK;
  for (int mi = 0; mi < 2; ++mi)
    for (int r = 0; r < 8; ++r) {
      const int i = gm + mi * 16 + r + hlf * 8;
      for (int ni = 0; ni < 4; ++ni)
        base[(size_t)i * NTOK + gn + ni * 16 + l15] =
            (_Float16)(acc[mi][ni][r] * SCALE);
    }
}

// ------------------------------------------------------------------ Softmax
__global__ void softmax_kernel(_Float16* __restrict__ attn) {
  const int t = threadIdx.x;
  _Float16* p = attn + (size_t)blockIdx.x * NTOK;

  float v[16];
  float m = -1e30f;
  for (int e = 0; e < 16; ++e) {
    v[e] = (float)p[t + e * 256];
    m = fmaxf(m, v[e]);
  }
  __shared__ float red[256];
  red[t] = m;
  __syncthreads();
  for (int o = 128; o > 0; o >>= 1) {
    if (t < o) red[t] = fmaxf(red[t], red[t + o]);
    __syncthreads();
  }
  m = red[0];
  __syncthreads();
  float s = 0.f;
  for (int e = 0; e < 16; ++e) { v[e] = __expf(v[e] - m); s += v[e]; }
  red[t] = s;
  __syncthreads();
  for (int o = 128; o > 0; o >>= 1) {
    if (t < o) red[t] += red[t + o];
    __syncthreads();
  }
  const float inv = 1.f / red[0];
  for (int e = 0; e < 16; ++e) p[t + e * 256] = (_Float16)(v[e] * inv);
}

// ------------------------------------------------------------------- AV GEMM
__global__ __launch_bounds__(256) void av_gemm(
    const _Float16* __restrict__ attn, const _Float16* __restrict__ vB,
    _Float16* __restrict__ outT) {
  const int b  = blockIdx.z;
  const int m0 = blockIdx.y * 128, n0 = blockIdx.x * 128;
  v8f z = {};
  v8f acc[2][4];
  for (int mi = 0; mi < 2; ++mi) for (int ni = 0; ni < 4; ++ni) acc[mi][ni] = z;

  gemm_core_128(attn + (size_t)b * NTOK * NTOK, NTOK,
                vB + (size_t)b * CDIM * NTOK, NTOK, NTOK, m0, n0, acc);

  const int lane = threadIdx.x & 31, w = threadIdx.x >> 5;
  const int hlf = lane >> 4, l15 = lane & 15;
  const int gm = m0 + (w >> 1) * 32, gn = n0 + (w & 1) * 64;
  _Float16* base = outT + (size_t)b * NTOK * CDIM;
  for (int mi = 0; mi < 2; ++mi)
    for (int r = 0; r < 8; ++r) {
      const int i = gm + mi * 16 + r + hlf * 8;
      for (int ni = 0; ni < 4; ++ni)
        base[(size_t)i * CDIM + gn + ni * 16 + l15] = (_Float16)acc[mi][ni][r];
    }
}

// ----------------------------------------------------------------- Proj GEMM
__global__ __launch_bounds__(256) void proj_gemm(
    const _Float16* __restrict__ Wp, const _Float16* __restrict__ outT,
    const float* __restrict__ projb, const float* __restrict__ x,
    float* __restrict__ y) {
  const int b  = blockIdx.z;
  const int m0 = blockIdx.y * 128, n0 = blockIdx.x * 128;
  v8f z = {};
  v8f acc[2][4];
  for (int mi = 0; mi < 2; ++mi) for (int ni = 0; ni < 4; ++ni) acc[mi][ni] = z;

  gemm_core_128(Wp, CDIM, outT + (size_t)b * NTOK * CDIM, CDIM, CDIM, m0, n0, acc);

  const int lane = threadIdx.x & 31, w = threadIdx.x >> 5;
  const int hlf = lane >> 4, l15 = lane & 15;
  const int gm = m0 + (w >> 1) * 32, gn = n0 + (w & 1) * 64;
  for (int mi = 0; mi < 2; ++mi)
    for (int r = 0; r < 8; ++r) {
      const int d = gm + mi * 16 + r + hlf * 8;
      const float bias = projb[d];
      for (int ni = 0; ni < 4; ++ni) {
        const size_t off = ((size_t)b * CDIM + d) * NTOK + gn + ni * 16 + l15;
        y[off] = x[off] + acc[mi][ni][r] + bias;
      }
    }
}

extern "C" void kernel_launch(void* const* d_in, const int* in_sizes, int n_in,
                              void* d_out, int out_size, void* d_ws, size_t ws_size,
                              hipStream_t stream) {
  const float* x     = (const float*)d_in[0];
  const float* gamma = (const float*)d_in[1];
  const float* beta  = (const float*)d_in[2];
  const float* qkvw  = (const float*)d_in[3];
  const float* qkvb  = (const float*)d_in[4];
  const float* projw = (const float*)d_in[5];
  const float* projb = (const float*)d_in[6];
  float* out = (float*)d_out;

  char* ws = (char*)d_ws;
  const size_t S = (size_t)BATCH * NTOK * CDIM * sizeof(_Float16);  // 8 MB
  _Float16* hT   = (_Float16*)(ws);
  _Float16* qT   = (_Float16*)(ws + 1 * S);
  _Float16* kT   = (_Float16*)(ws + 2 * S);
  _Float16* vB   = (_Float16*)(ws + 3 * S);
  _Float16* oT   = (_Float16*)(ws + 4 * S);
  _Float16* wq16 = (_Float16*)(ws + 5 * S);
  _Float16* wp16 = (_Float16*)(ws + 5 * S + (size_t)3 * CDIM * CDIM * sizeof(_Float16));
  _Float16* attn = (_Float16*)(ws + 5 * S + (size_t)4 * CDIM * CDIM * sizeof(_Float16));
  // total ws use: 5*8MB + 0.5MB + 128MB ~= 169 MB

  f32_to_f16<<<(3 * CDIM * CDIM + 255) / 256, 256, 0, stream>>>(qkvw, wq16,
                                                                3 * CDIM * CDIM);
  f32_to_f16<<<(CDIM * CDIM + 255) / 256, 256, 0, stream>>>(projw, wp16,
                                                            CDIM * CDIM);
  groupnorm_kernel<<<BATCH * NGRP, 256, 0, stream>>>(x, gamma, beta, hT);
  qkv_gemm<<<dim3(NTOK / 128, (3 * CDIM) / 128, BATCH), 256, 0, stream>>>(
      wq16, hT, qkvb, qT, kT, vB);
  scores_gemm<<<dim3(NTOK / 128, NTOK / 128, BATCH), 256, 0, stream>>>(qT, kT,
                                                                       attn);
  softmax_kernel<<<BATCH * NTOK, 256, 0, stream>>>(attn);
  av_gemm<<<dim3(CDIM / 128, NTOK / 128, BATCH), 256, 0, stream>>>(attn, vB, oT);
  proj_gemm<<<dim3(NTOK / 128, CDIM / 128, BATCH), 256, 0, stream>>>(wp16, oT,
                                                                     projb, x, out);
}
